// Custom_Linear_48395691491516
// MI455X (gfx1250) — 8-bit matrix operand order repaired and checked by reading, compiled, not run on hardware
//
#include <hip/hip_runtime.h>
#include <stdint.h>
#include <stddef.h>


typedef __attribute__((ext_vector_type(8))) int v8i;
typedef __attribute__((ext_vector_type(4))) int v4i_t;
typedef __attribute__((address_space(1))) v4i_t v4i_glob;
typedef __attribute__((address_space(3))) v4i_t v4i_lds;

#define M_TOT 8192
#define N_TOT 4096
#define K_TOT 4096
#define BM    128
#define BN    128
#define KB    64
#define LDSTR 80
#define NKSTEPS (K_TOT / KB)

#if __has_builtin(__builtin_amdgcn_global_load_async_to_lds_b128) && \
    __has_builtin(__builtin_amdgcn_s_wait_asynccnt)
#define USE_ASYNC_LDS 1
#else
#define USE_ASYNC_LDS 0
#endif

__device__ __forceinline__ unsigned pack4(int a, int b, int c, int d) {
    unsigned t01 = __builtin_amdgcn_perm((unsigned)b, (unsigned)a, 0x0c0c0400u);
    unsigned t23 = __builtin_amdgcn_perm((unsigned)d, (unsigned)c, 0x0c0c0400u);
    return __builtin_amdgcn_perm(t23, t01, 0x05040100u);
}

__device__ __forceinline__ uint4 pack16(const int4* g) {
    int4 d0 = g[0], d1 = g[1], d2 = g[2], d3 = g[3];
    uint4 p;
    p.x = pack4(d0.x, d0.y, d0.z, d0.w);
    p.y = pack4(d1.x, d1.y, d1.z, d1.w);
    p.z = pack4(d2.x, d2.y, d2.z, d2.w);
    p.w = pack4(d3.x, d3.y, d3.z, d3.w);
    return p;
}

__global__ __launch_bounds__(256)
void Custom_Linear_prepack_kernel(const int* __restrict__ src,
                                  uint8_t* __restrict__ dst, int n16)
{
    int i = blockIdx.x * blockDim.x + threadIdx.x;
    if (i < n16)
        ((uint4*)dst)[i] = pack16(((const int4*)src) + (size_t)i * 4);
}

template <bool PACKED>
__global__ __launch_bounds__(256)
void Custom_Linear_gemm_kernel(const void* __restrict__ Araw,
                               const void* __restrict__ Braw,
                               const float* __restrict__ scale_x,
                               const float* __restrict__ scale_w,
                               const float* __restrict__ scale_y,
                               float* __restrict__ out)
{
    __shared__ unsigned char sA[2][BM * LDSTR];
    __shared__ unsigned char sB[2][BN * LDSTR];

    const int tid  = threadIdx.x;
    const int lane = tid & 31;
    const int wave = tid >> 5;
    const int wm   = wave & 1;
    const int wn   = wave >> 1;
    const int half = lane >> 4;
    const int l16  = lane & 15;

    const int m0 = blockIdx.y * BM;
    const int n0 = blockIdx.x * BN;

    v8i acc[4][2] = {};

    const int row0  = tid >> 2;
    const int row1  = (tid + 256) >> 2;
    const int koff0 = (tid & 3) << 4;

    auto stage = [&](int k0, unsigned char* lA, unsigned char* lB) {
        #pragma unroll
        for (int it = 0; it < 2; ++it) {
            const int row = it ? row1 : row0;
            unsigned char* la = lA + row * LDSTR + koff0;
            unsigned char* lb = lB + row * LDSTR + koff0;
            if constexpr (PACKED) {
                const uint8_t* ga = (const uint8_t*)Araw + (size_t)(m0 + row) * K_TOT + k0 + koff0;
                const uint8_t* gb = (const uint8_t*)Braw + (size_t)(n0 + row) * K_TOT + k0 + koff0;
#if USE_ASYNC_LDS
                __builtin_amdgcn_global_load_async_to_lds_b128(
                    (v4i_glob*)const_cast<uint8_t*>(ga), (v4i_lds*)la, 0, 0);
                __builtin_amdgcn_global_load_async_to_lds_b128(
                    (v4i_glob*)const_cast<uint8_t*>(gb), (v4i_lds*)lb, 0, 0);
#else
                *(uint4*)la = *(const uint4*)ga;
                *(uint4*)lb = *(const uint4*)gb;
#endif
            } else {
                const int* xi = (const int*)Araw;
                const int* wi = (const int*)Braw;
                *(uint4*)la = pack16((const int4*)(xi + (size_t)(m0 + row) * K_TOT + k0 + koff0));
                *(uint4*)lb = pack16((const int4*)(wi + (size_t)(n0 + row) * K_TOT + k0 + koff0));
            }
        }
    };

    unsigned char* curA = sA[0];
    unsigned char* curB = sB[0];
    unsigned char* nxtA = sA[1];
    unsigned char* nxtB = sB[1];

    stage(0, curA, curB);
#if USE_ASYNC_LDS
    if constexpr (PACKED) __builtin_amdgcn_s_wait_asynccnt(0);
#endif
    __syncthreads();

    #pragma unroll 1
    for (int kt = 0; kt < NKSTEPS; ++kt) {
        if (kt + 1 < NKSTEPS) stage((kt + 1) * KB, nxtA, nxtB);

        v8i af[4];
        #pragma unroll
        for (int i = 0; i < 4; ++i) {
            const unsigned char* p = curA + (size_t)(wm * 64 + i * 16 + l16) * LDSTR + (half << 4);
            int2 q0 = *(const int2*)(p);
            int2 q1 = *(const int2*)(p + 8);
            int2 q2 = *(const int2*)(p + 32);
            int2 q3 = *(const int2*)(p + 40);
            v8i a;
            a[0] = q0.x; a[1] = q0.y; a[2] = q1.x; a[3] = q1.y;
            a[4] = q2.x; a[5] = q2.y; a[6] = q3.x; a[7] = q3.y;
            af[i] = a;
        }
        v8i bf[2];
        #pragma unroll
        for (int j = 0; j < 2; ++j) {
            const unsigned char* p = curB + (size_t)(wn * 32 + j * 16 + l16) * LDSTR + (half << 4);
            int4 q0 = *(const int4*)(p);
            int4 q1 = *(const int4*)(p + 32);
            v8i b;
            b[0] = q0.x; b[1] = q0.y; b[2] = q0.z; b[3] = q0.w;
            b[4] = q1.x; b[5] = q1.y; b[6] = q1.z; b[7] = q1.w;
            bf[j] = b;
        }

        #pragma unroll
        for (int i = 0; i < 4; ++i)
            #pragma unroll
            for (int j = 0; j < 2; ++j)
                acc[i][j] = __builtin_amdgcn_wmma_i32_16x16x64_iu8(
                     true, af[i],  true, bf[j],
                    acc[i][j],  false,  false);

        unsigned char* tA = curA; curA = nxtA; nxtA = tA;
        unsigned char* tB = curB; curB = nxtB; nxtB = tB;

#if USE_ASYNC_LDS
        if constexpr (PACKED) __builtin_amdgcn_s_wait_asynccnt(0);
#endif
        __syncthreads();
    }

    const float sxw = scale_x[0] * scale_w[0];
    #pragma unroll
    for (int j = 0; j < 2; ++j) {
        const int   col = n0 + wn * 32 + j * 16 + l16;
        const float rs  = sxw / scale_y[col];
        #pragma unroll
        for (int i = 0; i < 4; ++i) {
            const int rbase = m0 + wm * 64 + i * 16 + half * 8;
            #pragma unroll
            for (int e = 0; e < 8; ++e) {
                float f = (float)acc[i][j][e] * rs;
                f = rintf(f);
                f = fminf(fmaxf(f, -128.0f), 127.0f);
                out[(size_t)(rbase + e) * N_TOT + col] = f;
            }
        }
    }
}

__global__ void Custom_Linear_scaletail_kernel(const float* __restrict__ sy,
                                               float* __restrict__ out_tail)
{
    int i = blockIdx.x * blockDim.x + threadIdx.x;
    if (i < N_TOT) out_tail[i] = sy[i];
}

extern "C" void kernel_launch(void* const* d_in, const int* in_sizes, int n_in,
                              void* d_out, int out_size, void* d_ws, size_t ws_size,
                              hipStream_t stream) {
    (void)in_sizes; (void)n_in; (void)out_size;
    const int*   x  = (const int*)d_in[0];
    const int*   w  = (const int*)d_in[1];
    const float* sx = (const float*)d_in[2];
    const float* sw = (const float*)d_in[3];
    const float* sy = (const float*)d_in[4];
    float* out = (float*)d_out;

    const size_t xBytes = (size_t)M_TOT * K_TOT;
    const size_t wBytes = (size_t)N_TOT * K_TOT;
    dim3 grid(N_TOT / BN, M_TOT / BM);

    if (d_ws && ws_size >= xBytes + wBytes) {
        uint8_t* xp = (uint8_t*)d_ws;
        uint8_t* wp = xp + xBytes;
        const int nx16 = (int)(xBytes / 16);
        const int nw16 = (int)(wBytes / 16);
        Custom_Linear_prepack_kernel<<<(nx16 + 255) / 256, 256, 0, stream>>>(x, xp, nx16);
        Custom_Linear_prepack_kernel<<<(nw16 + 255) / 256, 256, 0, stream>>>(w, wp, nw16);
        Custom_Linear_gemm_kernel<true><<<grid, 256, 0, stream>>>(xp, wp, sx, sw, sy, out);
    } else {
        Custom_Linear_gemm_kernel<false><<<grid, 256, 0, stream>>>(x, w, sx, sw, sy, out);
    }
    Custom_Linear_scaletail_kernel<<<(N_TOT + 255) / 256, 256, 0, stream>>>(
        sy, out + (size_t)M_TOT * N_TOT);
}
